// LinearAttention_8358006358680
// MI455X (gfx1250) — matrix operand order repaired and checked by reading, not compiled, not run on hardware
//
#include <hip/hip_runtime.h>
#include <hip/hip_bf16.h>
#include <cstdint>
#include <cstddef>

// ---------------------------------------------------------------------------
// Causal linear attention (16 heads, d=1024, S=2048) for MI455X (gfx1250).
//  * Projections: v_wmma_f32_16x16x32_bf16, 32x64 tile/wave, double-buffered.
//  * Attention: CHUNKED linear attention (chunk=64) so the recurrence runs on
//    the matrix pipe too: per-chunk V^T K (WMMA) -> tiny per-head prefix sum
//    -> per-chunk Q@state + masked(QK^T)@V (WMMA, scores staged in LDS).
// ---------------------------------------------------------------------------

typedef __bf16 bf16_t;
typedef __attribute__((ext_vector_type(16))) __bf16 v16bf;
typedef __attribute__((ext_vector_type(2)))  __bf16 v2bf;
typedef __attribute__((ext_vector_type(8)))  float  v8f;

#define SEQ     2048
#define DMODEL  1024
#define NHEADS  16
#define DH      64
#define CH      64          // chunk length
#define NCH     (SEQ / CH)  // 32 chunks

// ---- workspace layout (bytes) ---------------------------------------------
#define MiB ((size_t)1 << 20)
static const size_t XB_OFF    = 0;                  // bf16 x [2048][1024]           4 MiB
static const size_t WQKV_OFF  = XB_OFF    + 4*MiB;  // bf16 [3*1024][1024]           6 MiB
static const size_t WO_OFF    = WQKV_OFF  + 6*MiB;  // bf16 [1024][1024]             2 MiB
static const size_t QBF_OFF   = WO_OFF    + 2*MiB;  // f32 q  [H][S][64]             8 MiB
static const size_t KBF_OFF   = QBF_OFF   + 8*MiB;  // f32 k  [H][S][64]             8 MiB
static const size_t Q16_OFF   = KBF_OFF   + 8*MiB;  // bf16 q [H][S][64]             4 MiB
static const size_t K16_OFF   = Q16_OFF   + 4*MiB;  // bf16 k [H][S][64]             4 MiB
static const size_t KT16_OFF  = K16_OFF   + 4*MiB;  // bf16 k^T [H][64][S]           4 MiB
static const size_t VT16_OFF  = KT16_OFF  + 4*MiB;  // bf16 v^T [H][64][S]           4 MiB
static const size_t GC_OFF    = VT16_OFF  + 4*MiB;  // f32 G_c [H][32][64][64]       8 MiB
static const size_t STPRE_OFF = GC_OFF    + 8*MiB;  // bf16 state prefix [H][32][64][64] 4 MiB
static const size_t K1PRE_OFF = STPRE_OFF + 4*MiB;  // f32 k1 prefix [H][32][64]   128 KiB
static const size_t AO_OFF    = K1PRE_OFF + 1*MiB;  // bf16 attn out [2048][1024]    4 MiB

__device__ __forceinline__ float softplusf(float x) {
    return (x > 20.0f) ? x : log1pf(__expf(x));
}

// ---- f32 -> bf16 cast ------------------------------------------------------
__global__ void cast_f32_to_bf16(const float* __restrict__ src,
                                 bf16_t* __restrict__ dst, int n) {
    int i = blockIdx.x * blockDim.x + threadIdx.x;
    if (i < n) dst[i] = (bf16_t)src[i];
}

// ---- WMMA fragment loads (ld = leading dimension in elements) --------------
// A: 16x32 bf16 tile, row-major source.
// Lanes 0-15: M=lane, K=0..15; lanes 16-31: M=lane-16, K=16..31.
__device__ __forceinline__ v16bf load_frag_a(const bf16_t* __restrict__ base,
                                             int row, int ld, int k0, int lane) {
    const bf16_t* p = base + (size_t)row * ld + k0;
    int kh = ((lane >> 4) << 4);           // 0 or 16
    v16bf a;
#pragma unroll
    for (int r = 0; r < 4; ++r) {
        v2bf lo = *(const v2bf*)(p + kh + 2 * r);
        v2bf hi = *(const v2bf*)(p + 8 + kh + 2 * r);
        a[2 * r]         = lo[0];
        a[2 * r + 1]     = lo[1];
        a[8 + 2 * r]     = hi[0];
        a[8 + 2 * r + 1] = hi[1];
    }
    return a;
}

// B: 32x16 bf16 tile with B[k][n] = W[n][k], W row-major [n][ld].
// Lanes 0-15: N=lane, K=0..15; lanes 16-31: N=lane-16, K=16..31.
__device__ __forceinline__ v16bf load_frag_b(const bf16_t* __restrict__ w,
                                             int n, int ld, int k0, int lane) {
    const bf16_t* p = w + (size_t)n * ld + k0 + ((lane >> 4) << 4);
    return *(const v16bf*)p;
}

// ---- shared 32(M)x64(N) x 1024(K) accumulation, double-buffered ------------
__device__ __forceinline__ void mm_accum(const bf16_t* __restrict__ Abase,
                                         const bf16_t* __restrict__ Wbase,
                                         int am0, int nl, int lane,
                                         v8f acc[2][4]) {
    v16bf a[2], b[4];
#pragma unroll
    for (int i = 0; i < 2; ++i) a[i] = load_frag_a(Abase, am0 + 16 * i, DMODEL, 0, lane);
#pragma unroll
    for (int t = 0; t < 4; ++t) b[t] = load_frag_b(Wbase, nl + 16 * t, DMODEL, 0, lane);

    for (int k0 = 0; k0 < DMODEL - 32; k0 += 32) {
        v16bf an[2], bn[4];
#pragma unroll
        for (int i = 0; i < 2; ++i)
            an[i] = load_frag_a(Abase, am0 + 16 * i, DMODEL, k0 + 32, lane);
#pragma unroll
        for (int t = 0; t < 4; ++t)
            bn[t] = load_frag_b(Wbase, nl + 16 * t, DMODEL, k0 + 32, lane);
        __builtin_prefetch(Abase + (size_t)am0 * DMODEL + ((k0 + 64) & (DMODEL - 1)), 0, 1);
#pragma unroll
        for (int i = 0; i < 2; ++i)
#pragma unroll
            for (int t = 0; t < 4; ++t)
                acc[i][t] = __builtin_amdgcn_wmma_f32_16x16x32_bf16(
                    false, a[i], false, b[t], (short)0, acc[i][t], false, false);
#pragma unroll
        for (int i = 0; i < 2; ++i) a[i] = an[i];
#pragma unroll
        for (int t = 0; t < 4; ++t) b[t] = bn[t];
    }
#pragma unroll
    for (int i = 0; i < 2; ++i)
#pragma unroll
        for (int t = 0; t < 4; ++t)
            acc[i][t] = __builtin_amdgcn_wmma_f32_16x16x32_bf16(
                false, a[i], false, b[t], (short)0, acc[i][t], false, false);
}

// ---- fused QKV projection with multi-layout epilogue ------------------------
__global__ void __launch_bounds__(256)
gemm_qkv(const bf16_t* __restrict__ xb, const bf16_t* __restrict__ wqkv,
         const float* __restrict__ bq, const float* __restrict__ bk,
         const float* __restrict__ bv, const float* __restrict__ beta,
         float* __restrict__ qbf, float* __restrict__ kbf,
         bf16_t* __restrict__ q16, bf16_t* __restrict__ k16,
         bf16_t* __restrict__ kt16, bf16_t* __restrict__ vt16) {
    const int lane = threadIdx.x & 31;
    const int wave = threadIdx.x >> 5;
    const int m0   = blockIdx.y * 256 + wave * 32;
    const int nblk = blockIdx.x * 64;
    const int am0  = m0 + (lane & 15);
    const int nl   = nblk + (lane & 15);

    v8f acc[2][4];
#pragma unroll
    for (int i = 0; i < 2; ++i)
#pragma unroll
        for (int t = 0; t < 4; ++t)
#pragma unroll
            for (int j = 0; j < 8; ++j) acc[i][t][j] = 0.0f;

    mm_accum(xb, wqkv, am0, nl, lane, acc);

#pragma unroll
    for (int i = 0; i < 2; ++i) {
#pragma unroll
        for (int t = 0; t < 4; ++t) {
            const int n_g  = nl + t * 16;
            const int proj = n_g >> 10;          // 0=Q,1=K,2=V
            const int d    = n_g & (DMODEL - 1);
            const int h    = d >> 6;
            const int e    = d & (DH - 1);
#pragma unroll
            for (int r = 0; r < 8; ++r) {
                const int s = m0 + i * 16 + r + ((lane >> 4) << 3);
                const size_t didx = (((size_t)h * SEQ) + s) * DH + e;   // [H][S][64]
                const size_t tidx = ((size_t)(h * DH + e)) * SEQ + s;   // [H][64][S]
                float v = acc[i][t][r];
                if (proj == 0) {
                    float sp = softplusf((v + bq[d]) / (8.0f * __expf(beta[h])));
                    qbf[didx] = sp;
                    q16[didx] = (bf16_t)sp;
                } else if (proj == 1) {
                    float sp = softplusf((v + bk[d]) / (8.0f * __expf(beta[h])));
                    kbf[didx]  = sp;
                    k16[didx]  = (bf16_t)sp;
                    kt16[tidx] = (bf16_t)sp;
                } else {
                    vt16[tidx] = (bf16_t)(v + bv[d]);
                }
            }
        }
    }
}

// ---- pass 1: per-chunk G_c[e][d] = sum_j v[j][e]*k[j][d]  (V^T K, WMMA) ----
__global__ void __launch_bounds__(256)
chunk_kv(const bf16_t* __restrict__ vt16, const bf16_t* __restrict__ kt16,
         float* __restrict__ gc) {
    const int h = blockIdx.y, ch = blockIdx.x;
    const int lane = threadIdx.x & 31, wave = threadIdx.x >> 5;
    const bf16_t* Abase = vt16 + (size_t)h * DH * SEQ;   // [e][s]
    const bf16_t* Bbase = kt16 + (size_t)h * DH * SEQ;   // [d][s]
    float* G = gc + (((size_t)h * NCH) + ch) * DH * DH;

#pragma unroll
    for (int tt = 0; tt < 2; ++tt) {
        const int tile = wave + tt * 8;          // 16 tiles over 8 waves
        const int e0 = (tile >> 2) * 16, d0 = (tile & 3) * 16;
        v8f acc;
#pragma unroll
        for (int j = 0; j < 8; ++j) acc[j] = 0.0f;
#pragma unroll
        for (int ks = 0; ks < 2; ++ks) {
            const int k0 = ch * CH + ks * 32;
            v16bf a = load_frag_a(Abase, e0 + (lane & 15), SEQ, k0, lane);
            v16bf b = load_frag_b(Bbase, d0 + (lane & 15), SEQ, k0, lane);
            acc = __builtin_amdgcn_wmma_f32_16x16x32_bf16(
                false, a, false, b, (short)0, acc, false, false);
        }
#pragma unroll
        for (int r = 0; r < 8; ++r) {
            const int e = e0 + r + ((lane >> 4) << 3);
            const int d = d0 + (lane & 15);
            G[e * DH + d] = acc[r];
        }
    }
}

// ---- pass 2: per-head exclusive prefix over chunk states -------------------
__global__ void __launch_bounds__(256)
chunk_prefix(const float* __restrict__ gc, const float* __restrict__ kbf,
             bf16_t* __restrict__ stpre, float* __restrict__ k1pre) {
    const int h = blockIdx.x;
    const int tid = threadIdx.x;

    float st[16];
#pragma unroll
    for (int i = 0; i < 16; ++i) st[i] = 0.0f;
    float k1 = 0.0f;

    for (int t = 0; t < NCH; ++t) {
        const size_t mbase = (((size_t)h * NCH) + t) * (DH * DH);
#pragma unroll
        for (int i = 0; i < 16; ++i) {
            stpre[mbase + tid * 16 + i] = (bf16_t)st[i];      // exclusive
            st[i] += gc[mbase + tid * 16 + i];
        }
        if (tid < DH) {
            k1pre[(((size_t)h * NCH) + t) * DH + tid] = k1;   // exclusive
            const float* kcol = kbf + (((size_t)h * SEQ) + t * CH) * DH + tid;
#pragma unroll
            for (int j = 0; j < CH; ++j) k1 += kcol[(size_t)j * DH];
        }
    }
}

// ---- pass 3: per-chunk output ----------------------------------------------
// S = QK^T masked (WMMA -> LDS bf16); den = q.k1pre + rowsum(S);
// num = Q @ ST_pre + S @ V  (both WMMA); out = num/den -> bf16 ao.
__global__ void __launch_bounds__(256)
chunk_attn(const bf16_t* __restrict__ q16, const bf16_t* __restrict__ k16,
           const bf16_t* __restrict__ vt16, const bf16_t* __restrict__ stpre,
           const float* __restrict__ qbf, const float* __restrict__ k1pre,
           bf16_t* __restrict__ ao) {
    const int h = blockIdx.y, ch = blockIdx.x;
    const int lane = threadIdx.x & 31, wave = threadIdx.x >> 5;
    __shared__ bf16_t Sm[CH * CH];
    __shared__ float dens[CH];

    const bf16_t* Qb = q16 + (size_t)h * SEQ * DH;              // [s][d]
    const bf16_t* Kb = k16 + (size_t)h * SEQ * DH;              // [s][d]
    const bf16_t* Vt = vt16 + (size_t)h * DH * SEQ;             // [e][s]
    const bf16_t* St = stpre + (((size_t)h * NCH) + ch) * (DH * DH); // [e][d]
    const int s0 = ch * CH;

    // --- masked scores into LDS ---
#pragma unroll
    for (int tt = 0; tt < 2; ++tt) {
        const int tile = wave + tt * 8;
        const int i0 = (tile >> 2) * 16, j0 = (tile & 3) * 16;
        v8f acc;
#pragma unroll
        for (int j = 0; j < 8; ++j) acc[j] = 0.0f;
#pragma unroll
        for (int ks = 0; ks < 2; ++ks) {
            v16bf a = load_frag_a(Qb, s0 + i0 + (lane & 15), DH, ks * 32, lane);
            v16bf b = load_frag_b(Kb, s0 + j0 + (lane & 15), DH, ks * 32, lane);
            acc = __builtin_amdgcn_wmma_f32_16x16x32_bf16(
                false, a, false, b, (short)0, acc, false, false);
        }
#pragma unroll
        for (int r = 0; r < 8; ++r) {
            const int i = i0 + r + ((lane >> 4) << 3);
            const int j = j0 + (lane & 15);
            Sm[i * CH + j] = (bf16_t)((j <= i) ? acc[r] : 0.0f);  // causal
        }
    }
    __syncthreads();

    // --- denominators ---
    if (threadIdx.x < CH) {
        const int i = threadIdx.x;
        float den = 0.0f;
        const float* qrow = qbf + (((size_t)h * SEQ) + s0 + i) * DH;
        const float* k1p  = k1pre + (((size_t)h * NCH) + ch) * DH;
#pragma unroll
        for (int d = 0; d < DH; ++d) den = fmaf(qrow[d], k1p[d], den);
#pragma unroll
        for (int j = 0; j < CH; ++j) den += (float)Sm[i * CH + j];
        dens[i] = den;
    }
    __syncthreads();

    // --- numerators: inter-chunk (Q @ state) + intra-chunk (Sm @ V) ---
#pragma unroll
    for (int tt = 0; tt < 2; ++tt) {
        const int tile = wave + tt * 8;
        const int i0 = (tile >> 2) * 16, e0 = (tile & 3) * 16;
        v8f acc;
#pragma unroll
        for (int j = 0; j < 8; ++j) acc[j] = 0.0f;
#pragma unroll
        for (int ks = 0; ks < 2; ++ks) {          // over d: Q @ ST_pre
            v16bf a = load_frag_a(Qb, s0 + i0 + (lane & 15), DH, ks * 32, lane);
            v16bf b = load_frag_b(St, e0 + (lane & 15), DH, ks * 32, lane);
            acc = __builtin_amdgcn_wmma_f32_16x16x32_bf16(
                false, a, false, b, (short)0, acc, false, false);
        }
#pragma unroll
        for (int ks = 0; ks < 2; ++ks) {          // over j: Sm(LDS) @ V
            v16bf a = load_frag_a((const bf16_t*)Sm, i0 + (lane & 15), CH, ks * 32, lane);
            v16bf b = load_frag_b(Vt, e0 + (lane & 15), SEQ, s0 + ks * 32, lane);
            acc = __builtin_amdgcn_wmma_f32_16x16x32_bf16(
                false, a, false, b, (short)0, acc, false, false);
        }
#pragma unroll
        for (int r = 0; r < 8; ++r) {
            const int i = i0 + r + ((lane >> 4) << 3);
            const int e = e0 + (lane & 15);
            ao[(size_t)(s0 + i) * DMODEL + h * DH + e] = (bf16_t)(acc[r] / dens[i]);
        }
    }
}

// ---- output projection: out = attn @ Wo^T + bo ----------------------------
__global__ void __launch_bounds__(256)
gemm_out(const bf16_t* __restrict__ ao, const bf16_t* __restrict__ wo,
         const float* __restrict__ bo, float* __restrict__ out) {
    const int lane = threadIdx.x & 31;
    const int wave = threadIdx.x >> 5;
    const int m0   = blockIdx.y * 256 + wave * 32;
    const int nblk = blockIdx.x * 64;
    const int am0  = m0 + (lane & 15);
    const int nl   = nblk + (lane & 15);

    v8f acc[2][4];
#pragma unroll
    for (int i = 0; i < 2; ++i)
#pragma unroll
        for (int t = 0; t < 4; ++t)
#pragma unroll
            for (int j = 0; j < 8; ++j) acc[i][t][j] = 0.0f;

    mm_accum(ao, wo, am0, nl, lane, acc);

#pragma unroll
    for (int i = 0; i < 2; ++i)
#pragma unroll
        for (int t = 0; t < 4; ++t) {
            const int n_g = nl + t * 16;
#pragma unroll
            for (int r = 0; r < 8; ++r) {
                const int m_g = m0 + i * 16 + r + ((lane >> 4) << 3);
                out[(size_t)m_g * DMODEL + n_g] = acc[i][t][r] + bo[n_g];
            }
        }
}

// ---------------------------------------------------------------------------
extern "C" void kernel_launch(void* const* d_in, const int* in_sizes, int n_in,
                              void* d_out, int out_size, void* d_ws, size_t ws_size,
                              hipStream_t stream) {
    (void)in_sizes; (void)n_in; (void)out_size; (void)ws_size;

    const float* x    = (const float*)d_in[0];
    const float* Wq   = (const float*)d_in[1];
    const float* bq   = (const float*)d_in[2];
    const float* Wk   = (const float*)d_in[3];
    const float* bk   = (const float*)d_in[4];
    const float* Wv   = (const float*)d_in[5];
    const float* bv   = (const float*)d_in[6];
    const float* Wo   = (const float*)d_in[7];
    const float* bo   = (const float*)d_in[8];
    const float* beta = (const float*)d_in[9];
    float* out = (float*)d_out;

    char* ws = (char*)d_ws;
    bf16_t* xb    = (bf16_t*)(ws + XB_OFF);
    bf16_t* wqkv  = (bf16_t*)(ws + WQKV_OFF);
    bf16_t* wob   = (bf16_t*)(ws + WO_OFF);
    float*  qbf   = (float*)(ws + QBF_OFF);
    float*  kbf   = (float*)(ws + KBF_OFF);
    bf16_t* q16   = (bf16_t*)(ws + Q16_OFF);
    bf16_t* k16   = (bf16_t*)(ws + K16_OFF);
    bf16_t* kt16  = (bf16_t*)(ws + KT16_OFF);
    bf16_t* vt16  = (bf16_t*)(ws + VT16_OFF);
    float*  gc    = (float*)(ws + GC_OFF);
    bf16_t* stpre = (bf16_t*)(ws + STPRE_OFF);
    float*  k1pre = (float*)(ws + K1PRE_OFF);
    bf16_t* aob   = (bf16_t*)(ws + AO_OFF);

    const int nx = SEQ * DMODEL;
    const int nw = DMODEL * DMODEL;
    cast_f32_to_bf16<<<(nx + 255) / 256, 256, 0, stream>>>(x, xb, nx);
    cast_f32_to_bf16<<<(nw + 255) / 256, 256, 0, stream>>>(Wq, wqkv, nw);
    cast_f32_to_bf16<<<(nw + 255) / 256, 256, 0, stream>>>(Wk, wqkv + (size_t)nw, nw);
    cast_f32_to_bf16<<<(nw + 255) / 256, 256, 0, stream>>>(Wv, wqkv + (size_t)2 * nw, nw);
    cast_f32_to_bf16<<<(nw + 255) / 256, 256, 0, stream>>>(Wo, wob, nw);

    dim3 g1(3 * DMODEL / 64, SEQ / 256);   // (48, 8)
    gemm_qkv<<<g1, 256, 0, stream>>>(xb, wqkv, bq, bk, bv, beta,
                                     qbf, kbf, q16, k16, kt16, vt16);

    dim3 gc2(NCH, NHEADS);                 // (32, 16)
    chunk_kv<<<gc2, 256, 0, stream>>>(vt16, kt16, gc);
    chunk_prefix<<<NHEADS, 256, 0, stream>>>(gc, kbf, stpre, k1pre);
    chunk_attn<<<gc2, 256, 0, stream>>>(q16, k16, vt16, stpre, qbf, k1pre, aob);

    dim3 g2(DMODEL / 64, SEQ / 256);       // (16, 8)
    gemm_out<<<g2, 256, 0, stream>>>(aob, wob, bo, out);
}
